// GPNN_HICO_32366873543034
// MI455X (gfx1250) — matrix operand order repaired and checked by reading, not compiled, not run on hardware
//
#include <hip/hip_runtime.h>
#include <hip/hip_bf16.h>
#include <math.h>

typedef __attribute__((ext_vector_type(16))) __bf16 v16bf;
typedef __attribute__((ext_vector_type(8)))  float  v8f;

union FragBF { v16bf v; int4 q[2]; };

// ---------------------------------------------------------------------------
// Generic GEMM: Y[M,N] = X[M,K] @ W.T + bias, W row-major [N][ldw] (+woff),
// bf16 WMMA with fp32 accumulate. Tile 128x64, 256 threads = 8 waves (4x2),
// each wave 32x32 = 2x2 fragments of 16x16. Double-buffered LDS staging.
// REQUIRES: M % 128 == 0, K % 32 == 0 (true for all call sites).
// ---------------------------------------------------------------------------
__global__ __launch_bounds__(256)
void gemm_bf16_kernel(const float* __restrict__ X, const float* __restrict__ W,
                      const float* __restrict__ bias, float* __restrict__ Y,
                      int M, int Nn, int K, int ldw, int woff)
{
    __shared__ __bf16 Als[2][128 * 32];
    __shared__ __bf16 Bls[2][64 * 32];

    const int tid  = threadIdx.x;
    const int lane = tid & 31;
    const int wid  = tid >> 5;
    const int wm   = wid >> 1;       // 0..3
    const int wn   = wid & 1;        // 0..1
    const int hi   = lane >> 4;      // 0/1
    const int l15  = lane & 15;

    const int m0 = blockIdx.x * 128;
    const int n0 = blockIdx.y * 64;

    // per-thread staging coordinates (fixed across K-steps)
    const int arow_[4] = { (0*256 + tid) >> 3, (1*256 + tid) >> 3,
                           (2*256 + tid) >> 3, (3*256 + tid) >> 3 };
    const int ac4      = (tid & 7) << 2;
    const int brow_[2] = { (0*256 + tid) >> 3, (1*256 + tid) >> 3 };

    v8f acc[2][2];
    #pragma unroll
    for (int i = 0; i < 2; ++i)
        #pragma unroll
        for (int j = 0; j < 2; ++j)
            acc[i][j] = v8f{0.f,0.f,0.f,0.f,0.f,0.f,0.f,0.f};

    const int ksteps = K >> 5;

    float4 ar[4], br[2];
    // ---- stage tile 0 ----
    {
        #pragma unroll
        for (int p = 0; p < 4; ++p)
            ar[p] = *(const float4*)(X + (size_t)(m0 + arow_[p]) * K + ac4);
        #pragma unroll
        for (int p = 0; p < 2; ++p) {
            int gn = n0 + brow_[p];
            br[p] = (gn < Nn) ? *(const float4*)(W + (size_t)gn * ldw + woff + ac4)
                              : make_float4(0.f, 0.f, 0.f, 0.f);
        }
        #pragma unroll
        for (int p = 0; p < 4; ++p) {
            __bf16* d = &Als[0][arow_[p] * 32 + ac4];
            d[0] = (__bf16)ar[p].x; d[1] = (__bf16)ar[p].y;
            d[2] = (__bf16)ar[p].z; d[3] = (__bf16)ar[p].w;
        }
        #pragma unroll
        for (int p = 0; p < 2; ++p) {
            __bf16* d = &Bls[0][brow_[p] * 32 + ac4];
            d[0] = (__bf16)br[p].x; d[1] = (__bf16)br[p].y;
            d[2] = (__bf16)br[p].z; d[3] = (__bf16)br[p].w;
        }
    }

    int cur = 0;
    for (int kt = 0; kt < ksteps; ++kt) {
        __syncthreads();
        const bool more = (kt + 1 < ksteps);
        if (more) {
            const int k1 = (kt + 1) << 5;
            #pragma unroll
            for (int p = 0; p < 4; ++p) {
                const float* src = X + (size_t)(m0 + arow_[p]) * K + k1 + ac4;
                ar[p] = *(const float4*)src;
                if (kt + 2 < ksteps) __builtin_prefetch(src + 32, 0, 1);
            }
            #pragma unroll
            for (int p = 0; p < 2; ++p) {
                int gn = n0 + brow_[p];
                br[p] = (gn < Nn)
                    ? *(const float4*)(W + (size_t)gn * ldw + woff + k1 + ac4)
                    : make_float4(0.f, 0.f, 0.f, 0.f);
            }
        }

        // ---- compute on buffer `cur` ----
        FragBF a[2], b[2];
        #pragma unroll
        for (int f = 0; f < 2; ++f) {
            const __bf16* pa = &Als[cur][(wm * 32 + f * 16 + l15) * 32 + hi * 8];
            a[f].q[0] = *(const int4*)pa;
            a[f].q[1] = *(const int4*)(pa + 16);
            const __bf16* pb = &Bls[cur][(wn * 32 + f * 16 + l15) * 32 + hi * 8];
            b[f].q[0] = *(const int4*)pb;
            b[f].q[1] = *(const int4*)(pb + 16);
        }
        #pragma unroll
        for (int fm = 0; fm < 2; ++fm)
            #pragma unroll
            for (int fn = 0; fn < 2; ++fn)
                acc[fm][fn] = __builtin_amdgcn_wmma_f32_16x16x32_bf16(
                    false, a[fm].v, false, b[fn].v, (short)0, acc[fm][fn], false, false);

        // ---- write tile kt+1 into the other buffer ----
        if (more) {
            const int nxt = cur ^ 1;
            #pragma unroll
            for (int p = 0; p < 4; ++p) {
                __bf16* d = &Als[nxt][arow_[p] * 32 + ac4];
                d[0] = (__bf16)ar[p].x; d[1] = (__bf16)ar[p].y;
                d[2] = (__bf16)ar[p].z; d[3] = (__bf16)ar[p].w;
            }
            #pragma unroll
            for (int p = 0; p < 2; ++p) {
                __bf16* d = &Bls[nxt][brow_[p] * 32 + ac4];
                d[0] = (__bf16)br[p].x; d[1] = (__bf16)br[p].y;
                d[2] = (__bf16)br[p].z; d[3] = (__bf16)br[p].w;
            }
        }
        cur ^= 1;
    }

    #pragma unroll
    for (int fm = 0; fm < 2; ++fm) {
        #pragma unroll
        for (int fn = 0; fn < 2; ++fn) {
            int gn = n0 + wn * 32 + fn * 16 + l15;
            if (gn >= Nn) continue;
            float bv = bias ? bias[gn] : 0.0f;
            #pragma unroll
            for (int j = 0; j < 8; ++j) {
                int gm = m0 + wm * 32 + fm * 16 + j + hi * 8;
                Y[(size_t)gm * Nn + gn] = acc[fm][fn][j] + bv;
            }
        }
    }
}

// ---------------------------------------------------------------------------
// Fused link MLP: pred[r] = relu(ES[r] @ W_l1.T + b_l1) . w_l2 + b_l2
// sig_out = sigmoid(pred); optionally write raw pred.
// Block: 64 rows, K=256 staged once in LDS; 8 waves each own 64 hidden cols.
// B-fragments for a hidden chunk are hoisted to registers and reused across
// all 4 row-fragments. transposed=1: row (b,v,w) gathered from m1[b,w,v].
// ---------------------------------------------------------------------------
__global__ __launch_bounds__(256)
void link_mlp_kernel(const float* __restrict__ ES, const __bf16* __restrict__ W1bf,
                     const float* __restrict__ b_l1, const float* __restrict__ W_l2,
                     const float* __restrict__ b_l2, float* __restrict__ sig_out,
                     float* __restrict__ pred_out, int transposed)
{
    __shared__ __bf16 Als[64 * 256];
    __shared__ float  rowacc[64];

    const int tid  = threadIdx.x;
    const int lane = tid & 31;
    const int wid  = tid >> 5;
    const int hi   = lane >> 4;
    const int l15  = lane & 15;
    const int m0   = blockIdx.x * 64;

    if (tid < 64) rowacc[tid] = 0.0f;

    // stage 64x256 A tile (fp32 -> bf16), with optional (v,w) transpose gather
    #pragma unroll
    for (int p = 0; p < 16; ++p) {
        int idx = p * 256 + tid;
        int row = idx >> 6;            // 64 float4 per row
        int c4  = (idx & 63) << 2;
        int r   = m0 + row;
        int src;
        if (transposed) {
            int b = r >> 14;
            int v = (r >> 7) & 127;
            int w = r & 127;
            src = (b << 14) + (w << 7) + v;
        } else {
            src = r;
        }
        float4 x = *(const float4*)(ES + (size_t)src * 256 + c4);
        __bf16* d = &Als[row * 256 + c4];
        d[0] = (__bf16)x.x; d[1] = (__bf16)x.y;
        d[2] = (__bf16)x.z; d[3] = (__bf16)x.w;
    }
    __syncthreads();

    const int hb = wid * 64;           // this wave's hidden-column base
    float ps[4][8];
    #pragma unroll
    for (int fm = 0; fm < 4; ++fm)
        #pragma unroll
        for (int j = 0; j < 8; ++j) ps[fm][j] = 0.f;

    #pragma unroll 1
    for (int fn = 0; fn < 4; ++fn) {
        const int ncol = hb + fn * 16 + l15;
        const __bf16* wrow = W1bf + (size_t)ncol * 256;
        // hoist all 8 K-step B fragments for this hidden chunk
        FragBF bfr[8];
        #pragma unroll
        for (int kt = 0; kt < 8; ++kt) {
            const __bf16* pb = wrow + kt * 32 + hi * 8;
            bfr[kt].q[0] = *(const int4*)pb;
            bfr[kt].q[1] = *(const int4*)(pb + 16);
        }
        const float bl1 = b_l1[ncol];
        const float w2  = W_l2[ncol];

        #pragma unroll 1
        for (int fm = 0; fm < 4; ++fm) {
            const __bf16* arow = &Als[(fm * 16 + l15) * 256];
            v8f acc0 = v8f{0.f,0.f,0.f,0.f,0.f,0.f,0.f,0.f};
            v8f acc1 = v8f{0.f,0.f,0.f,0.f,0.f,0.f,0.f,0.f};
            #pragma unroll
            for (int kt = 0; kt < 8; kt += 2) {
                FragBF a0, a1;
                const __bf16* pa0 = arow + kt * 32 + hi * 8;
                a0.q[0] = *(const int4*)pa0;  a0.q[1] = *(const int4*)(pa0 + 16);
                const __bf16* pa1 = pa0 + 32;
                a1.q[0] = *(const int4*)pa1;  a1.q[1] = *(const int4*)(pa1 + 16);
                acc0 = __builtin_amdgcn_wmma_f32_16x16x32_bf16(
                    false, a0.v, false, bfr[kt].v,     (short)0, acc0, false, false);
                acc1 = __builtin_amdgcn_wmma_f32_16x16x32_bf16(
                    false, a1.v, false, bfr[kt + 1].v, (short)0, acc1, false, false);
            }
            #pragma unroll
            for (int j = 0; j < 8; ++j) {
                float hval = acc0[j] + acc1[j] + bl1;
                hval = hval > 0.f ? hval : 0.f;
                ps[fm][j] += hval * w2;
            }
        }
    }

    // reduce across the 16 lanes (different n, same row), then one atomic each
    #pragma unroll
    for (int fm = 0; fm < 4; ++fm) {
        #pragma unroll
        for (int j = 0; j < 8; ++j) {
            float v = ps[fm][j];
            #pragma unroll
            for (int s = 1; s < 16; s <<= 1) v += __shfl_xor(v, s, 32);
            if (l15 == 0) atomicAdd(&rowacc[fm * 16 + j + hi * 8], v);
        }
    }
    __syncthreads();

    if (tid < 64) {
        float pred = rowacc[tid] + b_l2[0];
        int r = m0 + tid;
        if (pred_out) pred_out[r] = pred;
        sig_out[r] = 1.0f / (1.0f + __expf(-pred));
    }
}

// m1 = sig * relu(msg_h[b,w,:] + msg_e[b,v,w,:])
__global__ void msg_mul_kernel(const float* __restrict__ msg_e, const float* __restrict__ msg_h,
                               const float* __restrict__ sig, float* __restrict__ m_out,
                               long long total4)
{
    long long i = (long long)blockIdx.x * blockDim.x + threadIdx.x;
    const long long stride = (long long)gridDim.x * blockDim.x;
    for (; i < total4; i += stride) {
        long long e0  = i << 2;
        long long row = e0 >> 8;                 // (b*N + v)*N + w
        int c = (int)(e0 & 255);
        int w = (int)(row & 127);
        int b = (int)(row >> 14);
        float s = sig[row];
        float4 me = *(const float4*)(msg_e + e0);
        float4 mh = *(const float4*)(msg_h + (((size_t)(b * 128 + w)) << 8) + c);
        float4 o;
        o.x = s * fmaxf(me.x + mh.x, 0.f);
        o.y = s * fmaxf(me.y + mh.y, 0.f);
        o.z = s * fmaxf(me.z + mh.z, 0.f);
        o.w = s * fmaxf(me.w + mh.w, 0.f);
        *(float4*)(m_out + e0) = o;
    }
}

// m_sum[b,v,c] = sum_w sig[b,v,w] * relu(msg_h[b,w,c] + msg_e[b,v,w,c])
__global__ __launch_bounds__(256)
void msg_sum_kernel(const float* __restrict__ msg_e, const float* __restrict__ msg_h,
                    const float* __restrict__ sig, float* __restrict__ m_sum)
{
    const int bv = blockIdx.x;       // 0..1023
    const int c  = threadIdx.x;      // 0..255
    const int b  = bv >> 7;
    float acc = 0.f;
    for (int w = 0; w < 128; ++w) {
        float s  = sig[bv * 128 + w];
        float me = msg_e[(((size_t)(bv * 128 + w)) << 8) + c];
        float mh = msg_h[(((size_t)(b * 128 + w)) << 8) + c];
        acc += s * fmaxf(me + mh, 0.f);
    }
    m_sum[(size_t)bv * 256 + c] = acc;
}

// GRU gate fusion
__global__ void gru_kernel(const float* __restrict__ gi, const float* __restrict__ gh,
                           const float* __restrict__ h, float* __restrict__ h_new, int total)
{
    int i = blockIdx.x * blockDim.x + threadIdx.x;
    if (i >= total) return;
    int node = i >> 8;
    int c    = i & 255;
    const float* gir = gi + (size_t)node * 768;
    const float* ghr = gh + (size_t)node * 768;
    float r = 1.f / (1.f + __expf(-(gir[c]       + ghr[c])));
    float z = 1.f / (1.f + __expf(-(gir[256 + c] + ghr[256 + c])));
    float n = tanhf(gir[512 + c] + r * ghr[512 + c]);
    h_new[i] = (1.f - z) * n + z * h[i];
}

__global__ void cvt_bf16_kernel(const float* __restrict__ src, __bf16* __restrict__ dst, int n)
{
    int i = blockIdx.x * blockDim.x + threadIdx.x;
    if (i < n) dst[i] = (__bf16)src[i];
}

// ---------------------------------------------------------------------------
extern "C" void kernel_launch(void* const* d_in, const int* in_sizes, int n_in,
                              void* d_out, int out_size, void* d_ws, size_t ws_size,
                              hipStream_t stream)
{
    const float* edge_features = (const float*)d_in[0];
    const float* node_features = (const float*)d_in[1];
    const float* W_er = (const float*)d_in[6];
    const float* b_er = (const float*)d_in[7];
    const float* W_nr = (const float*)d_in[8];
    const float* b_nr = (const float*)d_in[9];
    const float* W_l1 = (const float*)d_in[10];
    const float* b_l1 = (const float*)d_in[11];
    const float* W_l2 = (const float*)d_in[12];
    const float* b_l2 = (const float*)d_in[13];
    const float* W_m  = (const float*)d_in[14];
    const float* b_m  = (const float*)d_in[15];
    const float* W_ih = (const float*)d_in[16];
    const float* b_ih = (const float*)d_in[17];
    const float* W_hh = (const float*)d_in[18];
    const float* b_hh = (const float*)d_in[19];
    const float* W_r  = (const float*)d_in[20];
    const float* b_r  = (const float*)d_in[21];

    const size_t EDGES = 8ull * 128 * 128;   // 131072
    const size_t EN    = EDGES * 256;        // 33,554,432

    float* ws      = (float*)d_ws;
    float* e_buf   = ws;                     // e, later reused for m1
    float* msg_e   = e_buf + EN;
    float* h       = msg_e + EN;
    float* msg_h   = h + 262144;
    float* sig     = msg_h + 262144;
    float* m_sum   = sig + 131072;
    float* gi      = m_sum + 262144;
    float* gh      = gi + 786432;
    float* h_new   = gh + 786432;
    __bf16* W1bf   = (__bf16*)(h_new + 262144);

    float* pred_adj    = (float*)d_out;
    float* pred_labels = (float*)d_out + EDGES;

    dim3 blk(256);

    // W_l1 -> bf16 (512x256)
    cvt_bf16_kernel<<<512, 256, 0, stream>>>(W_l1, W1bf, 512 * 256);

    // e = edge_features @ W_er.T + b_er       (M=131072, K=1024, N=256)
    gemm_bf16_kernel<<<dim3(1024, 4), blk, 0, stream>>>(edge_features, W_er, b_er, e_buf,
                                                        131072, 256, 1024, 1024, 0);
    // h = node_features @ W_nr.T + b_nr       (M=1024, K=1024, N=256)
    gemm_bf16_kernel<<<dim3(8, 4), blk, 0, stream>>>(node_features, W_nr, b_nr, h,
                                                     1024, 256, 1024, 1024, 0);
    // msg_h = h @ W_mh.T                      (W_m[:, :256], ldw=512)
    gemm_bf16_kernel<<<dim3(8, 4), blk, 0, stream>>>(h, W_m, nullptr, msg_h,
                                                     1024, 256, 256, 512, 0);
    // msg_e = e @ W_me.T + b_m                (W_m[:, 256:], ldw=512)
    gemm_bf16_kernel<<<dim3(1024, 4), blk, 0, stream>>>(e_buf, W_m, b_m, msg_e,
                                                        131072, 256, 256, 512, 256);

    // ---- propagation iteration 1 ----
    link_mlp_kernel<<<2048, blk, 0, stream>>>(e_buf, W1bf, b_l1, W_l2, b_l2,
                                              sig, nullptr, 0);
    msg_mul_kernel<<<4096, blk, 0, stream>>>(msg_e, msg_h, sig, e_buf,
                                             (long long)(EN / 4));   // m1 overwrites e
    // ---- propagation iteration 2 ----
    link_mlp_kernel<<<2048, blk, 0, stream>>>(e_buf, W1bf, b_l1, W_l2, b_l2,
                                              sig, pred_adj, 1);     // transposed m1
    msg_sum_kernel<<<1024, blk, 0, stream>>>(msg_e, msg_h, sig, m_sum);

    // GRU
    gemm_bf16_kernel<<<dim3(8, 12), blk, 0, stream>>>(m_sum, W_ih, b_ih, gi,
                                                      1024, 768, 256, 256, 0);
    gemm_bf16_kernel<<<dim3(8, 12), blk, 0, stream>>>(h, W_hh, b_hh, gh,
                                                      1024, 768, 256, 256, 0);
    gru_kernel<<<1024, 256, 0, stream>>>(gi, gh, h, h_new, 262144);

    // readout: pred_node_labels = h_new @ W_r.T + b_r  (N=600 -> 10 col tiles)
    gemm_bf16_kernel<<<dim3(8, 10), blk, 0, stream>>>(h_new, W_r, b_r, pred_labels,
                                                      1024, 600, 256, 256, 0);
}
